// BlockSoftmaxLinearHybrid_4784593567894
// MI455X (gfx1250) — matrix operand order repaired and checked by reading, not compiled, not run on hardware
//
#include <hip/hip_runtime.h>
#include <math.h>

// CDNA5 / gfx1250 — wave32, WMMA 16x16x32 bf16, fp32 accumulate.
typedef __attribute__((ext_vector_type(16))) __bf16 v16bf;
typedef __attribute__((ext_vector_type(8)))  float  v8f;

namespace {
constexpr int Bc = 2, Hc = 16, Lc = 4096, Dc = 128, Fc = 64, Sc = 64, NBc = 64;
// padded LDS row strides (elements)
constexpr int LQK = 136;   // bf16 rows of 128 (q, k, phi_q, W^T, S^T mirror)
constexpr int LVT = 72;    // bf16 rows of 64  (v^T, phi_k^T)
constexpr int LSS = 132;   // f32  rows of 128 (S state)
constexpr int LSC = 72;    // 64x64 scratch (scores / u) and attn
constexpr float SCALING = 0.08838834764831845f; // 128^-0.5
constexpr float EPSV = 1e-6f;
}

__device__ __forceinline__ v8f wmma_bf16(v16bf a, v16bf b, v8f c) {
  // D = A(16x32 bf16) x B(32x16 bf16) + C(16x16 f32)
  return __builtin_amdgcn_wmma_f32_16x16x32_bf16(false, a, false, b, (short)0, c, false, false);
}

// A fragment: base is [m][k] row-major bf16. lane&15 = m offset, lane>>4 selects K sub-half.
__device__ __forceinline__ v16bf load_a(const __bf16* base, int stride, int m0, int kbase, int lane) {
  const __bf16* p = base + (m0 + (lane & 15)) * stride + kbase + ((lane >> 4) << 4);
  v16bf a;
#pragma unroll
  for (int i = 0; i < 8; ++i) a[i] = p[i];          // K = kbase + 16*hi + 0..7
#pragma unroll
  for (int i = 0; i < 8; ++i) a[8 + i] = p[8 + i];  // K = kbase + 16*hi + 8..15
  return a;
}

// B fragment: base is [n][k] row-major bf16 (transposed so K is contiguous).
__device__ __forceinline__ v16bf load_b(const __bf16* base, int stride, int n0, int kbase, int lane) {
  const __bf16* p = base + (n0 + (lane & 15)) * stride + kbase + ((lane >> 4) << 4);
  v16bf b;
#pragma unroll
  for (int i = 0; i < 16; ++i) b[i] = p[i];         // K = kbase + 16*hi + i
  return b;
}

__device__ __forceinline__ v8f load_c(const float* base, int stride, int m0, int n0, int lane) {
  const int n  = n0 + (lane & 15);
  const int mm = m0 + ((lane >> 4) << 3);
  v8f c;
#pragma unroll
  for (int r = 0; r < 8; ++r) c[r] = base[(mm + r) * stride + n];
  return c;
}

__device__ __forceinline__ void store_c(float* base, int stride, int m0, int n0, int lane, v8f c) {
  const int n  = n0 + (lane & 15);
  const int mm = m0 + ((lane >> 4) << 3);
#pragma unroll
  for (int r = 0; r < 8; ++r) base[(mm + r) * stride + n] = c[r];
}

__global__ __launch_bounds__(512, 1)
void hedgehog_hybrid_kernel(const float* __restrict__ Q, const float* __restrict__ Kmat,
                            const float* __restrict__ V, const float* __restrict__ Wh,
                            const float* __restrict__ alphaPtr, float* __restrict__ Out) {
  __shared__ float  sS[128 * LSS];      // running state S (f32, persistent)
  __shared__ __bf16 sSbT[128 * LQK];    // bf16 mirror of S^T: [d][f] (B operand, refreshed per block)
  __shared__ float  sZ[128];            // running state Z
  __shared__ __bf16 sWT[64 * LQK];      // W^T: [f][d]  (B operand for u = x*W)
  __shared__ __bf16 sQ[64 * LQK];       // q block [s][d]
  __shared__ __bf16 sK[64 * LQK];       // k block [s][d]
  __shared__ __bf16 sVT[128 * LVT];     // v^T [d][s]   (B operand)
  __shared__ __bf16 sPhiQ[64 * LQK];    // phi_q [s][f2]
  __shared__ __bf16 sPhiKT[128 * LVT];  // phi_k^T [f2][s] (A operand for S update)
  __shared__ float  scr[64 * LSC];      // u / scores scratch
  __shared__ __bf16 sAttn[64 * LSC];    // softmax probs (A operand)
  __shared__ float  sDenom[64];

  const int tid  = threadIdx.x;
  const int lane = tid & 31;
  const int wave = tid >> 5;            // 16 waves
  const int bh   = blockIdx.x;
  const int h    = bh & (Hc - 1);

  const float wmix = 1.0f / (1.0f + __expf(-alphaPtr[0]));
  const float lmix = 1.0f - wmix;

  for (int g = tid; g < 128 * LSS; g += 512) sS[g] = 0.0f;
  for (int g = tid; g < 128 * LQK; g += 512) sSbT[g] = (__bf16)0.0f;
  if (tid < 128) sZ[tid] = 0.0f;
  for (int g = tid; g < Dc * Fc; g += 512) {           // stage W^T once
    int d = g >> 6, f = g & 63;
    sWT[f * LQK + d] = (__bf16)Wh[(h * Dc + d) * Fc + f];
  }
  __syncthreads();

  const size_t headBase = (size_t)bh * Lc * Dc;
  const float* qHead = Q + headBase;
  const float* kHead = Kmat + headBase;
  const float* vHead = V + headBase;
  float*       oHead = Out + headBase;

  for (int nb = 0; nb < NBc; ++nb) {
    // ---- phase 1: stage q,k,v block -> bf16 LDS ----
    const size_t blkOff = (size_t)nb * Sc * Dc;
    const float4* q4 = (const float4*)(qHead + blkOff);
    const float4* k4 = (const float4*)(kHead + blkOff);
    const float4* v4 = (const float4*)(vHead + blkOff);
    for (int g = tid; g < (Sc * Dc) / 4; g += 512) {
      const int s = g >> 5, d = (g & 31) << 2;
      float4 fq = q4[g];
      sQ[s*LQK+d+0]=(__bf16)fq.x; sQ[s*LQK+d+1]=(__bf16)fq.y;
      sQ[s*LQK+d+2]=(__bf16)fq.z; sQ[s*LQK+d+3]=(__bf16)fq.w;
      float4 fk = k4[g];
      sK[s*LQK+d+0]=(__bf16)fk.x; sK[s*LQK+d+1]=(__bf16)fk.y;
      sK[s*LQK+d+2]=(__bf16)fk.z; sK[s*LQK+d+3]=(__bf16)fk.w;
      float4 fv = v4[g];
      sVT[(d+0)*LVT+s]=(__bf16)fv.x; sVT[(d+1)*LVT+s]=(__bf16)fv.y;
      sVT[(d+2)*LVT+s]=(__bf16)fv.z; sVT[(d+3)*LVT+s]=(__bf16)fv.w;
    }
    if (nb + 1 < NBc) {  // global_prefetch_b8 next block (near-cache locality)
      const float* nq = qHead + blkOff + Sc * Dc;
      const float* nk = kHead + blkOff + Sc * Dc;
      const float* nv = vHead + blkOff + Sc * Dc;
      __builtin_prefetch(nq + tid * 16, 0, 3);
      __builtin_prefetch(nk + tid * 16, 0, 3);
      __builtin_prefetch(nv + tid * 16, 0, 3);
    }
    __syncthreads();

    // ---- phase 2: Uq = q * W  (64x64, K=128) : 16 tiles, one per wave ----
    {
      const int tm = (wave >> 2) << 4, tn = (wave & 3) << 4;
      v8f c = {};
#pragma unroll
      for (int kk = 0; kk < Dc; kk += 32)
        c = wmma_bf16(load_a(sQ, LQK, tm, kk, lane), load_b(sWT, LQK, tn, kk, lane), c);
      store_c(scr, LSC, tm, tn, lane, c);
    }
    __syncthreads();

    // ---- phase 3: phi_q = [softmax(u), softmax(-u)] over f ----
    if (tid < Sc) {
      const int s = tid;
      float mx = -3.4e38f, mn = 3.4e38f;
      for (int f = 0; f < Fc; ++f) { float v = scr[s*LSC+f]; mx = fmaxf(mx, v); mn = fminf(mn, v); }
      float sp = 0.f, sn = 0.f;
      for (int f = 0; f < Fc; ++f) { float v = scr[s*LSC+f]; sp += __expf(v - mx); sn += __expf(mn - v); }
      const float rp = 1.f / sp, rn = 1.f / sn;
      for (int f = 0; f < Fc; ++f) {
        float v = scr[s*LSC+f];
        sPhiQ[s*LQK + f]      = (__bf16)(__expf(v - mx) * rp);
        sPhiQ[s*LQK + Fc + f] = (__bf16)(__expf(mn - v) * rn);
      }
    }
    __syncthreads();

    // ---- phase 4: Uk = k * W ----
    {
      const int tm = (wave >> 2) << 4, tn = (wave & 3) << 4;
      v8f c = {};
#pragma unroll
      for (int kk = 0; kk < Dc; kk += 32)
        c = wmma_bf16(load_a(sK, LQK, tm, kk, lane), load_b(sWT, LQK, tn, kk, lane), c);
      store_c(scr, LSC, tm, tn, lane, c);
    }
    __syncthreads();

    // ---- phase 5: phi_k (transposed) ; denom = max(phi_q . Z, eps) ----
    if (tid < Sc) {
      const int s = tid;
      float mx = -3.4e38f, mn = 3.4e38f;
      for (int f = 0; f < Fc; ++f) { float v = scr[s*LSC+f]; mx = fmaxf(mx, v); mn = fminf(mn, v); }
      float sp = 0.f, sn = 0.f;
      for (int f = 0; f < Fc; ++f) { float v = scr[s*LSC+f]; sp += __expf(v - mx); sn += __expf(mn - v); }
      const float rp = 1.f / sp, rn = 1.f / sn;
      for (int f = 0; f < Fc; ++f) {
        float v = scr[s*LSC+f];
        sPhiKT[f * LVT + s]        = (__bf16)(__expf(v - mx) * rp);
        sPhiKT[(Fc + f) * LVT + s] = (__bf16)(__expf(mn - v) * rn);
      }
    } else if (tid < 128) {
      const int s = tid - 64;
      float acc = 0.f;
      for (int f = 0; f < 2 * Fc; ++f) acc += (float)sPhiQ[s*LQK + f] * sZ[f];
      sDenom[s] = fmaxf(acc, EPSV);
    }
    __syncthreads();

    // ---- phase 6: linear_out tiles (regs, shared B frags) + scores = q*k^T ----
    v8f lin0 = {}, lin1 = {};
    {
      const int tn  = (wave & 7) << 4;
      const int tm0 = (wave >> 3) << 4;   // 0 or 16
      const int tm1 = tm0 + 32;           // 32 or 48
#pragma unroll
      for (int kk = 0; kk < 2 * Fc; kk += 32) {
        const v16bf b = load_b(sSbT, LQK, tn, kk, lane);
        lin0 = wmma_bf16(load_a(sPhiQ, LQK, tm0, kk, lane), b, lin0);
        lin1 = wmma_bf16(load_a(sPhiQ, LQK, tm1, kk, lane), b, lin1);
      }
    }
    {
      const int tm = (wave >> 2) << 4, tn = (wave & 3) << 4;  // 64x64 scores
      v8f c = {};
#pragma unroll
      for (int kk = 0; kk < Dc; kk += 32)
        c = wmma_bf16(load_a(sQ, LQK, tm, kk, lane), load_b(sK, LQK, tn, kk, lane), c);
      store_c(scr, LSC, tm, tn, lane, c);
    }
    __syncthreads();

    // ---- phase 7: row softmax of scores ----
    if (tid < Sc) {
      const int s = tid;
      float mx = -3.4e38f;
      for (int t = 0; t < Sc; ++t) mx = fmaxf(mx, scr[s*LSC+t] * SCALING);
      float sum = 0.f;
      for (int t = 0; t < Sc; ++t) { float e = __expf(scr[s*LSC+t] * SCALING - mx); scr[s*LSC+t] = e; sum += e; }
      const float r = 1.f / sum;
      for (int t = 0; t < Sc; ++t) sAttn[s*LSC+t] = (__bf16)(scr[s*LSC+t] * r);
    }
    __syncthreads();

    // ---- phase 8: out = w*attn*v + (1-w)*lin/denom ; S += phi_k^T*v ; Z += colsum ----
    float* oBlk = oHead + blkOff;
    {
      const int tn  = (wave & 7) << 4;
      const int tm0 = (wave >> 3) << 4;
      const int tm1 = tm0 + 32;
      v8f c0 = {}, c1 = {};
#pragma unroll
      for (int kk = 0; kk < Sc; kk += 32) {
        const v16bf b = load_b(sVT, LVT, tn, kk, lane);   // shared across both tiles
        c0 = wmma_bf16(load_a(sAttn, LSC, tm0, kk, lane), b, c0);
        c1 = wmma_bf16(load_a(sAttn, LSC, tm1, kk, lane), b, c1);
      }
      const int n = tn + (lane & 15);
      const int r0 = tm0 + ((lane >> 4) << 3);
      const int r1 = tm1 + ((lane >> 4) << 3);
#pragma unroll
      for (int r = 0; r < 8; ++r) {
        oBlk[(size_t)(r0 + r) * Dc + n] = wmix * c0[r] + lmix * (lin0[r] / sDenom[r0 + r]);
        oBlk[(size_t)(r1 + r) * Dc + n] = wmix * c1[r] + lmix * (lin1[r] / sDenom[r1 + r]);
      }
    }
    {
      // S update: 64 tiles, 4 per wave sharing the A fragment (same tm, 4 consecutive tn)
      const int tm = (wave >> 1) << 4;
      const int tnB = (wave & 1) << 6;    // 0 or 64
      v8f c0 = load_c(sS, LSS, tm, tnB +  0, lane);
      v8f c1 = load_c(sS, LSS, tm, tnB + 16, lane);
      v8f c2 = load_c(sS, LSS, tm, tnB + 32, lane);
      v8f c3 = load_c(sS, LSS, tm, tnB + 48, lane);
#pragma unroll
      for (int kk = 0; kk < Sc; kk += 32) {
        const v16bf a = load_a(sPhiKT, LVT, tm, kk, lane);   // shared across 4 tiles
        c0 = wmma_bf16(a, load_b(sVT, LVT, tnB +  0, kk, lane), c0);
        c1 = wmma_bf16(a, load_b(sVT, LVT, tnB + 16, kk, lane), c1);
        c2 = wmma_bf16(a, load_b(sVT, LVT, tnB + 32, kk, lane), c2);
        c3 = wmma_bf16(a, load_b(sVT, LVT, tnB + 48, kk, lane), c3);
      }
      store_c(sS, LSS, tm, tnB +  0, lane, c0);
      store_c(sS, LSS, tm, tnB + 16, lane, c1);
      store_c(sS, LSS, tm, tnB + 32, lane, c2);
      store_c(sS, LSS, tm, tnB + 48, lane, c3);
    }
    if (tid < 128) {
      float z = 0.f;
      for (int s = 0; s < Sc; ++s) z += (float)sPhiKT[tid * LVT + s];
      sZ[tid] += z;
    }
    __syncthreads();

    // ---- phase 9: refresh bf16 transposed mirror of S for next block's linear GEMM ----
    for (int g = tid; g < Dc * Dc; g += 512) {
      const int d = g & 127, f = g >> 7;
      sSbT[d * LQK + f] = (__bf16)sS[f * LSS + d];
    }
    __syncthreads();
  }
}

extern "C" void kernel_launch(void* const* d_in, const int* in_sizes, int n_in,
                              void* d_out, int out_size, void* d_ws, size_t ws_size,
                              hipStream_t stream) {
  (void)in_sizes; (void)n_in; (void)out_size; (void)d_ws; (void)ws_size;
  const float* Q  = (const float*)d_in[0];
  const float* K  = (const float*)d_in[1];
  const float* V  = (const float*)d_in[2];
  const float* W  = (const float*)d_in[3];
  const float* al = (const float*)d_in[4];
  float* Out = (float*)d_out;
  hipLaunchKernelGGL(hedgehog_hybrid_kernel, dim3(Bc * Hc), dim3(512), 0, stream,
                     Q, K, V, W, al, Out);
}
